// LinearAttention_75505525064765
// MI455X (gfx1250) — matrix operand order repaired and checked by reading, not compiled, not run on hardware
//
#include <hip/hip_runtime.h>

// ---------------------------------------------------------------------------
// Galerkin linear attention for MI455X (gfx1250, wave32, WMMA 16x16x32 bf16).
//   b=4, n=8192, dim=256, heads=8, dim_head=64, inner=512
// All GEMMs on v_wmma_f32_16x16x32_bf16; fragments are ds_load_b128 pairs
// (A from [M][K] LDS, B from transposed [N][K] LDS). Big GEMMs use 128x64
// tiles: 8 WMMA per 12 b128 LDS loads per wave per k-chunk. K4 stages its
// tiles with global_load_async_to_lds_b128 (ASYNCcnt).
// ---------------------------------------------------------------------------

typedef __attribute__((ext_vector_type(16))) __bf16 v16bf;
typedef __attribute__((ext_vector_type(8)))  float  v8f;

#define B_BATCH   4
#define N_SEQ     8192
#define DIM       256
#define HEADS     8
#define DHEAD     64
#define INNER     512                 // HEADS * DHEAD
#define ROWS      (B_BATCH * N_SEQ)   // 32768
#define KSTRIDE   48                  // LDS stride for K=32 tiles (96B, 16B mult)

__device__ inline v8f wmma_bf16(v16bf a, v16bf b, v8f c) {
  return __builtin_amdgcn_wmma_f32_16x16x32_bf16(false, a, false, b,
                                                 (short)0, c, false, false);
}

union V16U { v16bf v; uint4 q[2]; };
union PK8  { uint4 q; __bf16 e[8]; };
union PK4  { unsigned long long u; __bf16 e[4]; };

// ---- fragment loaders: 2x ds_load_b128 each -------------------------------
// A 16x32 from [M][K] LDS: lane m=lane&15; elements = row[kh..kh+7],row[kh+16..kh+23]
__device__ inline v16bf load_A_frag(const __bf16* lds, int stride,
                                    int mBase, int kBase, int lane) {
  const __bf16* row = lds + (mBase + (lane & 15)) * stride + kBase + ((lane >> 4) * 8);
  V16U u;
  u.q[0] = *(const uint4*)(row);
  u.q[1] = *(const uint4*)(row + 16);
  return u.v;
}
// B 32x16 from transposed [N][K] LDS: lane n=lane&15; elements = row[kh..kh+7],row[kh+16..kh+23]
__device__ inline v16bf load_B_frag(const __bf16* ldsT, int stride,
                                    int nBase, int kBase, int lane) {
  const __bf16* row = ldsT + (nBase + (lane & 15)) * stride + kBase + ((lane >> 4) * 8);
  V16U u;
  u.q[0] = *(const uint4*)(row);
  u.q[1] = *(const uint4*)(row + 16);
  return u.v;
}

// ---- staging helpers (128-thread workgroups) ------------------------------
// f32 global -> bf16 LDS row-major; cols mult of 4, 8B packed LDS stores
__device__ inline void stage_f32_bf16(__bf16* lds, int ldsStride,
                                      const float* __restrict__ g, int ldg,
                                      int rows, int cols, int tid) {
  int total = rows * cols;
  for (int i = tid * 4; i < total; i += 128 * 4) {
    int r = i / cols, c = i % cols;
    const float4 f = *(const float4*)(g + (size_t)r * ldg + c);
    PK4 p;
    p.e[0] = (__bf16)f.x; p.e[1] = (__bf16)f.y;
    p.e[2] = (__bf16)f.z; p.e[3] = (__bf16)f.w;
    *(unsigned long long*)(lds + r * ldsStride + c) = p.u;
  }
}
// bf16 global -> bf16 LDS row-major, 16B both sides
__device__ inline void stage_bf16(__bf16* lds, int ldsStride,
                                  const __bf16* __restrict__ g, int ldg,
                                  int rows, int cols, int tid) {
  int total = rows * cols;
  for (int i = tid * 8; i < total; i += 128 * 8) {
    int r = i / cols, c = i % cols;
    *(uint4*)(lds + r * ldsStride + c) = *(const uint4*)(g + (size_t)r * ldg + c);
  }
}
// bf16 global [rows][cols] -> LDS transposed [cols][ldsStride]
__device__ inline void stage_bf16_T(__bf16* lds, int ldsStride,
                                    const __bf16* __restrict__ g, int ldg,
                                    int rows, int cols, int tid) {
  int total = rows * cols;
  for (int i = tid * 8; i < total; i += 128 * 8) {
    int r = i / cols, c = i % cols;
    PK8 u;
    u.q = *(const uint4*)(g + (size_t)r * ldg + c);
#pragma unroll
    for (int j = 0; j < 8; ++j) lds[(c + j) * ldsStride + r] = u.e[j];
  }
}
// f32 global [rows][cols] -> bf16 LDS transposed [cols][ldsStride]
__device__ inline void stage_f32_bf16_T(__bf16* lds, int ldsStride,
                                        const float* __restrict__ g, int ldg,
                                        int rows, int cols, int tid) {
  int total = rows * cols;
  for (int i = tid * 4; i < total; i += 128 * 4) {
    int r = i / cols, c = i % cols;
    float4 f = *(const float4*)(g + (size_t)r * ldg + c);
    lds[(c + 0) * ldsStride + r] = (__bf16)f.x;
    lds[(c + 1) * ldsStride + r] = (__bf16)f.y;
    lds[(c + 2) * ldsStride + r] = (__bf16)f.z;
    lds[(c + 3) * ldsStride + r] = (__bf16)f.w;
  }
}
// bf16 global -> LDS row-major via CDNA5 async-to-LDS DMA (ASYNCcnt)
__device__ inline void stage_bf16_async(__bf16* lds, int ldsStride,
                                        const __bf16* __restrict__ g, int ldg,
                                        int rows, int cols, int tid) {
  int total = rows * cols;
  for (int i = tid * 8; i < total; i += 128 * 8) {
    int r = i / cols, c = i % cols;
    unsigned int ldsOff = (unsigned int)(uintptr_t)(lds + r * ldsStride + c);
    unsigned long long ga = (unsigned long long)(uintptr_t)(g + (size_t)r * ldg + c);
    asm volatile("global_load_async_to_lds_b128 %0, %1, off"
                 :: "v"(ldsOff), "v"(ga) : "memory");
  }
}
__device__ inline void wait_async0() {
  asm volatile("s_wait_asynccnt 0x0" ::: "memory");
}

// ---------------------------------------------------------------------------
__global__ void zero_f32(float* p, int n) {
  int i = blockIdx.x * blockDim.x + threadIdx.x;
  if (i < n) p[i] = 0.0f;
}
// pre-transpose w_qkv (f32 [256][1536]) -> bf16 [1536][256]; tiny, once.
__global__ void transpose_wqkv(const float* __restrict__ w, __bf16* __restrict__ wT) {
  int i = blockIdx.x * 256 + threadIdx.x;
  if (i < 1536 * 256) {
    int n = i >> 8, k = i & 255;
    wT[i] = (__bf16)w[(size_t)k * 1536 + n];
  }
}

// ---------------------------------------------------------------------------
// K1: qkv = x @ w_qkv, fused instance-norm on k/v. 128x64 tile, 4 waves,
// each wave = 32-row strip (2 A frags x 4 B frags = 8 WMMA / k-chunk).
// grid (ROWS/128=256, 24), block 128.
__global__ __launch_bounds__(128)
void qkv_gemm_norm(const float* __restrict__ x, const __bf16* __restrict__ wqkvT,
                   __bf16* __restrict__ qb, __bf16* __restrict__ kb,
                   __bf16* __restrict__ vb) {
  __shared__ __attribute__((aligned(16))) __bf16 aL[128 * KSTRIDE];
  __shared__ __attribute__((aligned(16))) __bf16 bT[64 * KSTRIDE];
  __shared__ __attribute__((aligned(16))) float  cL[128 * 68];
  int tid = threadIdx.x, lane = tid & 31, wave = tid >> 5;
  int tm = blockIdx.x, tn = blockIdx.y;
  const float*  aG  = x + (size_t)tm * 128 * DIM;
  const __bf16* bGT = wqkvT + (size_t)tn * 64 * DIM;     // [n][k] rows

  v8f acc[2][4] = {};
  for (int kc = 0; kc < DIM; kc += 32) {
    __syncthreads();
    stage_f32_bf16(aL, KSTRIDE, aG + kc, DIM, 128, 32, tid);
    stage_bf16(bT, KSTRIDE, bGT + kc, DIM, 64, 32, tid); // already [N][K]
    __syncthreads();
    v16bf a0 = load_A_frag(aL, KSTRIDE, wave * 32, 0, lane);
    v16bf a1 = load_A_frag(aL, KSTRIDE, wave * 32 + 16, 0, lane);
#pragma unroll
    for (int t = 0; t < 4; ++t) {
      v16bf bf = load_B_frag(bT, KSTRIDE, t * 16, 0, lane);
      acc[0][t] = wmma_bf16(a0, bf, acc[0][t]);
      acc[1][t] = wmma_bf16(a1, bf, acc[1][t]);
    }
  }

  __syncthreads();
#pragma unroll
  for (int ma = 0; ma < 2; ++ma)
#pragma unroll
    for (int t = 0; t < 4; ++t)
#pragma unroll
      for (int r = 0; r < 8; ++r)
        cL[(wave * 32 + ma * 16 + r + ((lane >> 4) * 8)) * 68 +
           t * 16 + (lane & 15)] = acc[ma][t][r];
  __syncthreads();

  {
    size_t row = (size_t)tm * 128 + tid;
    const float* crow = &cL[tid * 68];
    int cat = tn >> 3;                 // 0:q 1:k 2:v
    int colBase = (tn & 7) * 64;       // whole head in this tile
    __bf16* dst = (cat == 0 ? qb : (cat == 1 ? kb : vb)) + row * INNER + colBase;
    float buf[64];
    float s = 0.f, ss = 0.f;
#pragma unroll
    for (int c4 = 0; c4 < 16; ++c4) {
      float4 f = *(const float4*)(crow + c4 * 4);
      buf[c4 * 4 + 0] = f.x; buf[c4 * 4 + 1] = f.y;
      buf[c4 * 4 + 2] = f.z; buf[c4 * 4 + 3] = f.w;
      s += f.x + f.y + f.z + f.w;
      ss += f.x * f.x + f.y * f.y + f.z * f.z + f.w * f.w;
    }
    float mu = 0.f, inv = 1.f;
    if (cat != 0) {
      mu = s * (1.0f / 64.0f);
      float var = ss * (1.0f / 64.0f) - mu * mu;
      inv = rsqrtf(var + 1e-5f);
    }
#pragma unroll
    for (int c8 = 0; c8 < 8; ++c8) {
      PK8 p;
#pragma unroll
      for (int j = 0; j < 8; ++j) p.e[j] = (__bf16)((buf[c8 * 8 + j] - mu) * inv);
      *(uint4*)(dst + c8 * 8) = p.q;
    }
  }
}

// ---------------------------------------------------------------------------
// K2: dots[b,h] += k^T v ; both tiles staged transposed into [headdim][n] LDS.
// grid (32 bh, 64 splits), block 128.
__global__ __launch_bounds__(128)
void kv_dots(const __bf16* __restrict__ kb, const __bf16* __restrict__ vb,
             float* __restrict__ dots) {
  __shared__ __attribute__((aligned(16))) __bf16 kT[64 * KSTRIDE];
  __shared__ __attribute__((aligned(16))) __bf16 vT[64 * KSTRIDE];
  int tid = threadIdx.x, lane = tid & 31, wave = tid >> 5;
  int bh = blockIdx.x, split = blockIdx.y;
  int b = bh >> 3, h = bh & 7;
  size_t rowBase = (size_t)b * N_SEQ + (size_t)split * 128;
  const __bf16* kG = kb + rowBase * INNER + h * 64;
  const __bf16* vG = vb + rowBase * INNER + h * 64;

  v8f acc[4] = {};
  for (int nc = 0; nc < 128; nc += 32) {
    __syncthreads();
    stage_bf16_T(kT, KSTRIDE, kG + (size_t)nc * INNER, INNER, 32, 64, tid);
    stage_bf16_T(vT, KSTRIDE, vG + (size_t)nc * INNER, INNER, 32, 64, tid);
    __syncthreads();
    v16bf a = load_A_frag(kT, KSTRIDE, wave * 16, 0, lane);  // A=k^T: [d][n]
#pragma unroll
    for (int t = 0; t < 4; ++t)
      acc[t] = wmma_bf16(a, load_B_frag(vT, KSTRIDE, t * 16, 0, lane), acc[t]);
  }

  float* dbase = dots + (size_t)bh * 4096;
#pragma unroll
  for (int t = 0; t < 4; ++t)
#pragma unroll
    for (int r = 0; r < 8; ++r) {
      int m = wave * 16 + r + ((lane >> 4) * 8);
      int n = t * 16 + (lane & 15);
      unsafeAtomicAdd(&dbase[m * 64 + n], acc[t][r]);  // global_atomic_add_f32
    }
}

// ---------------------------------------------------------------------------
// K3: W2^T[b, c, h*64+d] = sum_e (dots[b,h,d,e]/n) * w_out[h*64+e, c]
// grid 32 (bh), block 128. Emits W2 transposed so K4's B staging is direct.
__global__ __launch_bounds__(128)
void w2_kernel(const float* __restrict__ dots, const float* __restrict__ wout,
               __bf16* __restrict__ w2T) {
  __shared__ __attribute__((aligned(16))) __bf16 dL[64 * 72];
  __shared__ __attribute__((aligned(16))) __bf16 wLT[256 * 72];
  int tid = threadIdx.x, lane = tid & 31, wave = tid >> 5;
  int bh = blockIdx.x, b = bh >> 3, h = bh & 7;

  const float* dsrc = dots + (size_t)bh * 4096;
  for (int i = tid; i < 4096; i += 128)
    dL[(i >> 6) * 72 + (i & 63)] = (__bf16)(dsrc[i] * (1.0f / (float)N_SEQ));
  stage_f32_bf16_T(wLT, 72, wout + (size_t)h * 64 * DIM, DIM, 64, 256, tid);
  __syncthreads();

  v16bf a0 = load_A_frag(dL, 72, wave * 16, 0, lane);
  v16bf a1 = load_A_frag(dL, 72, wave * 16, 32, lane);
  __bf16* dst = w2T + (size_t)b * DIM * INNER;     // [c=256][inner=512]
#pragma unroll
  for (int nt = 0; nt < 16; ++nt) {
    v8f acc = {};
    acc = wmma_bf16(a0, load_B_frag(wLT, 72, nt * 16, 0, lane), acc);
    acc = wmma_bf16(a1, load_B_frag(wLT, 72, nt * 16, 32, lane), acc);
#pragma unroll
    for (int r = 0; r < 8; ++r) {
      int m = wave * 16 + r + ((lane >> 4) * 8);     // d
      int n = nt * 16 + (lane & 15);                 // c
      dst[(size_t)n * INNER + h * 64 + m] = (__bf16)acc[r];
    }
  }
}

// ---------------------------------------------------------------------------
// K4: out[b] = q[b] @ W2[b] + b_out. 128x64 tile, async-to-LDS staged.
// grid (64, 4, 4), block 128.
__global__ __launch_bounds__(128)
void out_gemm(const __bf16* __restrict__ qb, const __bf16* __restrict__ w2T,
              const float* __restrict__ bout, float* __restrict__ out) {
  __shared__ __attribute__((aligned(16))) __bf16 aL[128 * KSTRIDE];
  __shared__ __attribute__((aligned(16))) __bf16 bT[64 * KSTRIDE];
  int tid = threadIdx.x, lane = tid & 31, wave = tid >> 5;
  int tm = blockIdx.x, tn = blockIdx.y, b = blockIdx.z;
  const __bf16* aG  = qb + ((size_t)b * N_SEQ + (size_t)tm * 128) * INNER;
  const __bf16* bGT = w2T + (size_t)b * DIM * INNER + (size_t)tn * 64 * INNER;

  v8f acc[2][4] = {};
  for (int kc = 0; kc < INNER; kc += 32) {
    __syncthreads();
    stage_bf16_async(aL, KSTRIDE, aG + kc, INNER, 128, 32, tid);
    stage_bf16_async(bT, KSTRIDE, bGT + kc, INNER, 64, 32, tid);
    wait_async0();
    __syncthreads();
    v16bf a0 = load_A_frag(aL, KSTRIDE, wave * 32, 0, lane);
    v16bf a1 = load_A_frag(aL, KSTRIDE, wave * 32 + 16, 0, lane);
#pragma unroll
    for (int t = 0; t < 4; ++t) {
      v16bf bf = load_B_frag(bT, KSTRIDE, t * 16, 0, lane);
      acc[0][t] = wmma_bf16(a0, bf, acc[0][t]);
      acc[1][t] = wmma_bf16(a1, bf, acc[1][t]);
    }
  }

  float* obase = out + ((size_t)b * N_SEQ + (size_t)tm * 128) * DIM + tn * 64;
#pragma unroll
  for (int ma = 0; ma < 2; ++ma)
#pragma unroll
    for (int t = 0; t < 4; ++t)
#pragma unroll
      for (int r = 0; r < 8; ++r) {
        int m = wave * 32 + ma * 16 + r + ((lane >> 4) * 8);
        int n = t * 16 + (lane & 15);
        obase[(size_t)m * DIM + n] = acc[ma][t][r] + bout[tn * 64 + n];
      }
}

// ---------------------------------------------------------------------------
extern "C" void kernel_launch(void* const* d_in, const int* in_sizes, int n_in,
                              void* d_out, int out_size, void* d_ws, size_t ws_size,
                              hipStream_t stream) {
  const float* x    = (const float*)d_in[0];   // [4, 8192, 256]
  const float* wqkv = (const float*)d_in[1];   // [256, 1536]
  const float* wout = (const float*)d_in[2];   // [512, 256]
  const float* bout = (const float*)d_in[3];   // [256]
  float* out = (float*)d_out;                  // [4, 8192, 256] f32

  char* ws = (char*)d_ws;
  const size_t QKV_B = (size_t)ROWS * INNER * sizeof(__bf16);    // 32 MiB each
  __bf16* qb    = (__bf16*)(ws);
  __bf16* kb    = (__bf16*)(ws + QKV_B);
  __bf16* vb    = (__bf16*)(ws + 2 * QKV_B);
  float*  dots  = (float*)(ws + 3 * QKV_B);                      // 512 KiB
  __bf16* w2T   = (__bf16*)(ws + 3 * QKV_B + 524288);            // 1 MiB
  __bf16* wqkvT = (__bf16*)(ws + 3 * QKV_B + 524288 + 1048576);  // 768 KiB

  zero_f32<<<(B_BATCH * HEADS * 64 * 64 + 255) / 256, 256, 0, stream>>>(
      dots, B_BATCH * HEADS * 64 * 64);
  transpose_wqkv<<<(1536 * 256 + 255) / 256, 256, 0, stream>>>(wqkv, wqkvT);
  qkv_gemm_norm<<<dim3(ROWS / 128, 24), 128, 0, stream>>>(x, wqkvT, qb, kb, vb);
  kv_dots<<<dim3(B_BATCH * HEADS, 64), 128, 0, stream>>>(kb, vb, dots);
  w2_kernel<<<B_BATCH * HEADS, 128, 0, stream>>>(dots, wout, w2T);
  out_gemm<<<dim3(N_SEQ / 128, DIM / 64, B_BATCH), 128, 0, stream>>>(qb, w2T, bout, out);
}
